// seq2seq_20315195310140
// MI455X (gfx1250) — compile-verified, matrix operand order checked by reading, not run on hardware
//
#include <hip/hip_runtime.h>

// ---------------------------------------------------------------------------
// Types for CDNA5 WMMA (wave32): bf16 16x16x32, fp32 accumulate
// ---------------------------------------------------------------------------
typedef __bf16 bf16_t;
typedef __attribute__((ext_vector_type(16))) __bf16 v16bf;
typedef __attribute__((ext_vector_type(8)))  float  v8f;
typedef __attribute__((ext_vector_type(4)))  float  f32x4;
typedef __attribute__((ext_vector_type(4)))  int    i32x4;

union Frag {
    v16bf v;
    f32x4 q[2];
};

#define BQ   256   // batch
#define UQ   512   // hidden units
#define NQ   2048  // 4*U gate width
#define TENC 336
#define HOR  48

#define AS1 __attribute__((address_space(1)))
#define AS3 __attribute__((address_space(3)))

#if __has_builtin(__builtin_amdgcn_global_load_async_to_lds_b128)
#define HAVE_ASYNC_LDS 1
#else
#define HAVE_ASYNC_LDS 0
#endif

__device__ __forceinline__ float sigmoidf_(float x) {
    return 1.0f / (1.0f + __expf(-x));
}

// Copy one 16-byte chunk global -> LDS (async DMA if available)
__device__ __forceinline__ void copy16_g2l(const bf16_t* g, bf16_t* l) {
#if HAVE_ASYNC_LDS
    __builtin_amdgcn_global_load_async_to_lds_b128(
        (AS1 i32x4*)(AS1 void*)(void*)g,
        (AS3 i32x4*)(AS3 void*)(void*)l, 0, 0);
#else
    *(f32x4*)l = *(const f32x4*)g;
#endif
}

__device__ __forceinline__ void wait_async_copies() {
#if HAVE_ASYNC_LDS
#if __has_builtin(__builtin_amdgcn_s_wait_asynccnt)
    __builtin_amdgcn_s_wait_asynccnt(0);
#else
    asm volatile("s_wait_asynccnt 0" ::: "memory");
#endif
#endif
}

// ---- fragment helpers ------------------------------------------------------
__device__ __forceinline__ void load_bfrags(Frag* b, const bf16_t* __restrict__ Wt,
                                            int K, int n0, int mr, int koff)
{
#pragma unroll
    for (int j = 0; j < 4; ++j) {
        const bf16_t* bp = Wt + (size_t)(n0 + j * 16 + mr) * K + koff;
        b[j].q[0] = *(const f32x4*)(bp);
        b[j].q[1] = *(const f32x4*)(bp + 16);
    }
}

__device__ __forceinline__ void load_afrag(Frag& a, const bf16_t* ldsA,
                                           int K, int mr, int koff)
{
    const bf16_t* ap = ldsA + (size_t)mr * K + koff;
    a.q[0] = *(const f32x4*)(ap);
    a.q[1] = *(const f32x4*)(ap + 16);
}

__device__ __forceinline__ void wmma4(v8f* acc, const Frag& a, const Frag* b)
{
#pragma unroll
    for (int j = 0; j < 4; ++j)
        acc[j] = __builtin_amdgcn_wmma_f32_16x16x32_bf16(
            false, a.v, false, b[j].v, (short)0, acc[j], false, false);
}

// ---------------------------------------------------------------------------
// One 16(M) x 64(N) strip of z += A(16xK, LDS) @ Wt (transposed [N][K],
// global/L2).  Ping-pong double buffering with two disjoint register sets:
// no array rotation, no v_mov copies between WMMAs, one stage of lookahead.
// ---------------------------------------------------------------------------
__device__ __forceinline__ void gemm_accum(v8f* acc, const bf16_t* ldsA, int K,
                                           const bf16_t* __restrict__ Wt,
                                           int n0, int mr, int half)
{
    const int khalf = half * 8;
    Frag b0[4], b1[4], a0, a1;
    load_bfrags(b0, Wt, K, n0, mr, khalf);
    load_afrag(a0, ldsA, K, mr, khalf);
    for (int k = 0; k < K; k += 64) {
        const bool has1 = (k + 32) < K;
        if (has1) {
            load_bfrags(b1, Wt, K, n0, mr, k + 32 + khalf);
            load_afrag(a1, ldsA, K, mr, k + 32 + khalf);
        }
        wmma4(acc, a0, b0);
        if (k + 64 < K) {
            load_bfrags(b0, Wt, K, n0, mr, k + 64 + khalf);
            load_afrag(a0, ldsA, K, mr, k + 64 + khalf);
        }
        if (has1) wmma4(acc, a1, b1);
    }
}

// ---------------------------------------------------------------------------
// Fused LSTM step:  z = x @ Wx + h @ Wh + b ; gates ; c,h update
// Grid: (16, 8) -> (m-tile of 16 rows, u-tile of 64 units)
// Block: 128 threads = 4 waves; wave w computes gate w of the 16x64 tile.
// ---------------------------------------------------------------------------
__global__ __launch_bounds__(128)
void lstm_step_wmma(const bf16_t* __restrict__ xbf, int Kx,
                    const bf16_t* __restrict__ Wxt,
                    const bf16_t* __restrict__ hbf_in,
                    const bf16_t* __restrict__ Wht,
                    const float*  __restrict__ bias,
                    float*        __restrict__ c,
                    float*        __restrict__ hf_out,
                    bf16_t*       __restrict__ hbf_out)
{
    __shared__ bf16_t ldsX[16 * UQ];    // x strip (only 16*Kx used), 16 KB
    __shared__ bf16_t ldsH[16 * UQ];    // h strip, 16 KB
    __shared__ float  zt[4 * 16 * 64];  // z gate tile, 16 KB

    const int lane = threadIdx.x & 31;
    const int wave = threadIdx.x >> 5;        // gate id 0..3
    const int m0   = blockIdx.x * 16;         // batch-row base
    const int u0   = blockIdx.y * 64;         // hidden-unit base
    const int half = lane >> 4;               // K-half select per ISA layout
    const int mr   = lane & 15;               // M (A) / N (B) within tile
    const int n0   = wave * UQ + u0;          // z column base of this gate

    // ---- stage A strips (contiguous rows) into LDS via async DMA ----------
    const bf16_t* gx = xbf + (size_t)m0 * Kx;
    const bf16_t* gh = hbf_in + (size_t)m0 * UQ;
    const int cx = 2 * Kx;                     // 16B chunks in 16 x Kx strip
    for (int cid = threadIdx.x; cid < cx; cid += 128)
        copy16_g2l(gx + cid * 8, ldsX + cid * 8);
    for (int cid = threadIdx.x; cid < 1024; cid += 128)
        copy16_g2l(gh + cid * 8, ldsH + cid * 8);
    wait_async_copies();
    __syncthreads();

    // ---- GEMM: four 16x16 fp32 accumulators per wave -----------------------
    v8f acc[4] = {};
    gemm_accum(acc, ldsX, Kx, Wxt, n0, mr, half);
    gemm_accum(acc, ldsH, UQ, Wht, n0, mr, half);

    // ---- spill gate tile to LDS (C layout: VGPR r -> M = r + 8*half, N = mr)
#pragma unroll
    for (int j = 0; j < 4; ++j) {
#pragma unroll
        for (int r = 0; r < 8; ++r) {
            const int ml = r + half * 8;
            const int nl = j * 16 + mr;
            zt[wave * 1024 + ml * 64 + nl] = acc[j][r];
        }
    }
    __syncthreads();

    // ---- elementwise gates + cell update (8 elements / thread) -------------
    const int e0 = threadIdx.x * 8;
#pragma unroll
    for (int q = 0; q < 8; ++q) {
        const int e  = e0 + q;
        const int ml = e >> 6;
        const int ul = e & 63;
        const int u  = u0 + ul;
        const int row = m0 + ml;
        const int le = ml * 64 + ul;

        const float zi = zt[0 * 1024 + le] + bias[u];
        const float zf = zt[1 * 1024 + le] + bias[512  + u];
        const float zg = zt[2 * 1024 + le] + bias[1024 + u];
        const float zo = zt[3 * 1024 + le] + bias[1536 + u];

        const float ig = sigmoidf_(zi);
        const float fg = sigmoidf_(zf);
        const float gg = tanhf(zg);
        const float og = sigmoidf_(zo);

        const size_t idx = (size_t)row * UQ + u;
        const float cn = fg * c[idx] + ig * gg;
        const float hn = og * tanhf(cn);
        c[idx]       = cn;
        hf_out[idx]  = hn;
        hbf_out[idx] = (bf16_t)hn;
    }
}

// ---------------------------------------------------------------------------
// Convert+transpose weights: fp32 W[K][2048] -> bf16 Wt[2048][Kp] (zero pad)
// ---------------------------------------------------------------------------
__global__ void conv_wT(const float* __restrict__ W, bf16_t* __restrict__ Wt,
                        int K, int Kp, int total)
{
    const int idx = blockIdx.x * blockDim.x + threadIdx.x;
    if (idx >= total) return;
    const int n = idx / Kp;
    const int k = idx % Kp;
    Wt[idx] = (k < K) ? (bf16_t)W[(size_t)k * NQ + n] : (bf16_t)0.0f;
}

// fp32 encoder input [B][T][16] -> bf16 [T][B][32] (padded K)
__global__ void conv_xenc(const float* __restrict__ x, bf16_t* __restrict__ xe)
{
    const int idx = blockIdx.x * blockDim.x + threadIdx.x;
    if (idx >= TENC * BQ * 32) return;
    const int t = idx / (BQ * 32);
    const int r = idx % (BQ * 32);
    const int m = r >> 5;
    const int k = r & 31;
    xe[idx] = (k < 16) ? (bf16_t)x[((size_t)m * TENC + t) * 16 + k]
                       : (bf16_t)0.0f;
}

// Zero states; seed decoder x buffer with decoder_inputs in column 0
__global__ void init_state(float* c0, float* c1,
                           bf16_t* h0a, bf16_t* h0b, bf16_t* h1a, bf16_t* h1b,
                           const float* __restrict__ dec_in,
                           bf16_t* __restrict__ xdec)
{
    const int i = blockIdx.x * blockDim.x + threadIdx.x;
    if (i < BQ * UQ) {
        c0[i] = 0.0f; c1[i] = 0.0f;
        h0a[i] = (bf16_t)0.0f; h0b[i] = (bf16_t)0.0f;
        h1a[i] = (bf16_t)0.0f; h1b[i] = (bf16_t)0.0f;
    }
    if (i < BQ * 32) {
        const int m = i >> 5, k = i & 31;
        xdec[i] = (k == 0) ? (bf16_t)dec_in[m] : (bf16_t)0.0f;
    }
}

// y = h1 @ Wd + bd ; write output and feed back into decoder x buffer
__global__ void proj_out(const float* __restrict__ h1f,
                         const float* __restrict__ Wd,
                         const float* __restrict__ bd,
                         float* __restrict__ out, int step,
                         bf16_t* __restrict__ xdec)
{
    const int m = threadIdx.x;   // 256 threads
    float s = 0.0f;
#pragma unroll 8
    for (int k = 0; k < UQ; ++k) s += h1f[(size_t)m * UQ + k] * Wd[k];
    const float y = s + bd[0];
    out[(size_t)m * HOR + step] = y;
    xdec[m * 32] = (bf16_t)y;
}

// ---------------------------------------------------------------------------
extern "C" void kernel_launch(void* const* d_in, const int* in_sizes, int n_in,
                              void* d_out, int out_size, void* d_ws, size_t ws_size,
                              hipStream_t stream)
{
    const float* enc_in  = (const float*)d_in[0];
    const float* dec_in  = (const float*)d_in[1];
    const float* eWx0    = (const float*)d_in[2];
    const float* eWh0    = (const float*)d_in[3];
    const float* eb0     = (const float*)d_in[4];
    const float* eWx1    = (const float*)d_in[5];
    const float* eWh1    = (const float*)d_in[6];
    const float* eb1     = (const float*)d_in[7];
    const float* dWx0    = (const float*)d_in[8];
    const float* dWh0    = (const float*)d_in[9];
    const float* db0     = (const float*)d_in[10];
    const float* dWx1    = (const float*)d_in[11];
    const float* dWh1    = (const float*)d_in[12];
    const float* db1     = (const float*)d_in[13];
    const float* Wd      = (const float*)d_in[14];
    const float* bd      = (const float*)d_in[15];
    float* out = (float*)d_out;

    // ---- workspace carve-out (~20 MB) -------------------------------------
    char* ws = (char*)d_ws;
    size_t off = 0;
    auto carve = [&](size_t bytes) -> char* {
        char* p = ws + off;
        off = (off + bytes + 255) & ~(size_t)255;
        return p;
    };
    bf16_t* eWx0t = (bf16_t*)carve((size_t)NQ * 32  * 2);
    bf16_t* eWh0t = (bf16_t*)carve((size_t)NQ * UQ  * 2);
    bf16_t* eWx1t = (bf16_t*)carve((size_t)NQ * UQ  * 2);
    bf16_t* eWh1t = (bf16_t*)carve((size_t)NQ * UQ  * 2);
    bf16_t* dWx0t = (bf16_t*)carve((size_t)NQ * 32  * 2);
    bf16_t* dWh0t = (bf16_t*)carve((size_t)NQ * UQ  * 2);
    bf16_t* dWx1t = (bf16_t*)carve((size_t)NQ * UQ  * 2);
    bf16_t* dWh1t = (bf16_t*)carve((size_t)NQ * UQ  * 2);
    bf16_t* xe    = (bf16_t*)carve((size_t)TENC * BQ * 32 * 2);
    bf16_t* h0bf[2] = { (bf16_t*)carve((size_t)BQ * UQ * 2),
                        (bf16_t*)carve((size_t)BQ * UQ * 2) };
    bf16_t* h1bf[2] = { (bf16_t*)carve((size_t)BQ * UQ * 2),
                        (bf16_t*)carve((size_t)BQ * UQ * 2) };
    float* c0   = (float*)carve((size_t)BQ * UQ * 4);
    float* c1   = (float*)carve((size_t)BQ * UQ * 4);
    float* h0f  = (float*)carve((size_t)BQ * UQ * 4);
    float* h1f  = (float*)carve((size_t)BQ * UQ * 4);
    bf16_t* xdec = (bf16_t*)carve((size_t)BQ * 32 * 2);
    (void)ws_size; (void)in_sizes; (void)n_in; (void)out_size;

    // ---- one-time-per-call conversions ------------------------------------
    const int tSmall = NQ * 32,  gSmall = (tSmall + 255) / 256;
    const int tBig   = NQ * UQ,  gBig   = (tBig   + 255) / 256;
    conv_wT<<<gSmall, 256, 0, stream>>>(eWx0, eWx0t, 16,  32,  tSmall);
    conv_wT<<<gBig,   256, 0, stream>>>(eWh0, eWh0t, UQ,  UQ,  tBig);
    conv_wT<<<gBig,   256, 0, stream>>>(eWx1, eWx1t, UQ,  UQ,  tBig);
    conv_wT<<<gBig,   256, 0, stream>>>(eWh1, eWh1t, UQ,  UQ,  tBig);
    conv_wT<<<gSmall, 256, 0, stream>>>(dWx0, dWx0t, 1,   32,  tSmall);
    conv_wT<<<gBig,   256, 0, stream>>>(dWh0, dWh0t, UQ,  UQ,  tBig);
    conv_wT<<<gBig,   256, 0, stream>>>(dWx1, dWx1t, UQ,  UQ,  tBig);
    conv_wT<<<gBig,   256, 0, stream>>>(dWh1, dWh1t, UQ,  UQ,  tBig);
    conv_xenc<<<(TENC * BQ * 32 + 255) / 256, 256, 0, stream>>>(enc_in, xe);
    init_state<<<(BQ * UQ + 255) / 256, 256, 0, stream>>>(
        c0, c1, h0bf[0], h0bf[1], h1bf[0], h1bf[1], dec_in, xdec);

    // ---- recurrent sweep ---------------------------------------------------
    dim3 grid(16, 8), blk(128);
    int p0 = 0, p1 = 0;
    for (int t = 0; t < TENC; ++t) {
        lstm_step_wmma<<<grid, blk, 0, stream>>>(
            xe + (size_t)t * BQ * 32, 32, eWx0t,
            h0bf[p0], eWh0t, eb0, c0, h0f, h0bf[p0 ^ 1]);
        p0 ^= 1;
        lstm_step_wmma<<<grid, blk, 0, stream>>>(
            h0bf[p0], UQ, eWx1t,
            h1bf[p1], eWh1t, eb1, c1, h1f, h1bf[p1 ^ 1]);
        p1 ^= 1;
    }
    for (int s = 0; s < HOR; ++s) {
        lstm_step_wmma<<<grid, blk, 0, stream>>>(
            xdec, 32, dWx0t,
            h0bf[p0], dWh0t, db0, c0, h0f, h0bf[p0 ^ 1]);
        p0 ^= 1;
        lstm_step_wmma<<<grid, blk, 0, stream>>>(
            h0bf[p0], UQ, dWx1t,
            h1bf[p1], dWh1t, db1, c1, h1f, h1bf[p1 ^ 1]);
        p1 ^= 1;
        proj_out<<<1, 256, 0, stream>>>(h1f, Wd, bd, out, s, xdec);
    }
}
